// MSAttention_60825326846648
// MI455X (gfx1250) — compile-verified, matrix operand order checked by reading, not run on hardware
//
#include <hip/hip_runtime.h>

typedef __bf16 bf16_t;
typedef bf16_t v16bf __attribute__((ext_vector_type(16)));
typedef float  v8f   __attribute__((ext_vector_type(8)));

// ---------------------------------------------------------------------------
// Convert W_qkv[:,512:768] (fp32, row-major 256x768) -> bf16 in WMMA-B fragment
// layout: [ks(0..7)][nt(0..15)][lane(0..31)][j(0..15)]
//   lane<16 : column N = nt*16+lane,   K = ks*32 + {0..7,16..23}[j]
//   lane>=16: column N = nt*16+lane-16,K = ks*32 + {8..15,24..31}[j]
// ---------------------------------------------------------------------------
__global__ __launch_bounds__(32)
void convert_w_kernel(const float* __restrict__ W0, const float* __restrict__ W1,
                      bf16_t* __restrict__ Wb0, bf16_t* __restrict__ Wb1) {
    const int g    = blockIdx.x;          // 0..255
    const int s    = g >> 7;              // scale select
    const int rem  = g & 127;
    const int ks   = rem >> 4;            // K-step 0..7
    const int nt   = rem & 15;            // N-tile 0..15
    const int lane = threadIdx.x;         // 0..31 (wave32)

    const float* W  = s ? W1 : W0;
    bf16_t*      Wb = s ? Wb1 : Wb0;

    const int N  = nt * 16 + (lane & 15);
    const int kb = (lane < 16) ? 0 : 8;
    bf16_t* dst  = Wb + (((ks * 16 + nt) * 32 + lane) << 4);

#pragma unroll
    for (int j = 0; j < 16; ++j) {
        const int kl = ((j < 8) ? j : (j + 8)) + kb;   // {0..7,16..23} or {8..15,24..31}
        const int k  = ks * 32 + kl;
        dst[j] = (bf16_t)W[k * 768 + 512 + N];
    }
}

// ---------------------------------------------------------------------------
// Per-block V projection:  one workgroup = one 4x4 spatial block (16 tokens).
// 4 waves, each wave computes a 16(M) x 64(N) slice of V = X @ Wv via
// v_wmma_f32_16x16x32_bf16 (8 K-steps x 4 N-tiles = 32 WMMAs/wave).
// Writes optional full V (scale 0) and always the 16-token block sum S.
// ---------------------------------------------------------------------------
__global__ __launch_bounds__(128)
void gemm_v_block_kernel(const float* __restrict__ X, const bf16_t* __restrict__ Wb,
                         float* __restrict__ Vout, float* __restrict__ Ssum,
                         const int Hdim) {
    const int  G   = Hdim >> 2;                      // block grid dim (8 or 32)
    const int  blk = blockIdx.x;                     // (b*G + hb)*G + wb
    const int  wbi = blk % G;
    const int  hbi = (blk / G) % G;
    const int  b   = blk / (G * G);
    const long baseTok = ((long)b * Hdim + hbi * 4) * Hdim + wbi * 4;

    __shared__ bf16_t sA[16 * 264];                  // 16 rows x 256 K, padded stride

    // ---- stage A-tile: 16 token rows x 256 features, fp32 -> bf16 ----
    {
        const int  t   = threadIdx.x;                // 0..127
        const int  r   = t >> 3;                     // token row 0..15
        const int  q   = t & 7;
        const long tok = baseTok + (long)(r >> 2) * Hdim + (r & 3);
        const float* src = X + tok * 256;
        bf16_t*      dst = sA + r * 264;
#pragma unroll
        for (int c = 0; c < 8; ++c) {
            const int e = (q + 8 * c) * 4;           // element index, 16B-aligned
            const float4 f = *(const float4*)(src + e);
            union { bf16_t h[4]; uint2 u; } p;
            p.h[0] = (bf16_t)f.x; p.h[1] = (bf16_t)f.y;
            p.h[2] = (bf16_t)f.z; p.h[3] = (bf16_t)f.w;
            *(uint2*)(dst + e) = p.u;
        }
    }
    __syncthreads();

    const int lane  = threadIdx.x & 31;
    const int wave  = threadIdx.x >> 5;              // 0..3
    const int nbase = wave * 64;
    const int arow  = lane & 15;
    const int abase = (lane < 16) ? 0 : 8;

    v8f acc[4] = {};

#pragma unroll
    for (int ks = 0; ks < 8; ++ks) {
        union { v16bf v; uint4 u[2]; } a;
        const bf16_t* ap = sA + arow * 264 + ks * 32 + abase;
        a.u[0] = *(const uint4*)(ap);                // K {0..7}  (+8 for hi half)
        a.u[1] = *(const uint4*)(ap + 16);           // K {16..23}(+8 for hi half)
#pragma unroll
        for (int j = 0; j < 4; ++j) {
            const int   nt = (nbase >> 4) + j;
            const v16bf bv = *(const v16bf*)(Wb + (((ks * 16 + nt) * 32 + lane) << 4));
            acc[j] = __builtin_amdgcn_wmma_f32_16x16x32_bf16(
                false, a.v, false, bv, (short)0, acc[j], false, false);
        }
    }

    const int mofs = (lane < 16) ? 0 : 8;            // D layout: VGPR r -> M = r + mofs

    if (Vout) {
#pragma unroll
        for (int j = 0; j < 4; ++j) {
            const int ncol = nbase + j * 16 + (lane & 15);
#pragma unroll
            for (int r = 0; r < 8; ++r) {
                const int  m   = r + mofs;
                const long tok = baseTok + (long)(m >> 2) * Hdim + (m & 3);
                Vout[tok * 256 + ncol] = acc[j][r];
            }
        }
    }

    // ---- 16-token block sum: reduce accumulator M-rows, then wave halves ----
#pragma unroll
    for (int j = 0; j < 4; ++j) {
        float s = 0.f;
#pragma unroll
        for (int r = 0; r < 8; ++r) s += acc[j][r];
        s += __shfl_xor(s, 16, 32);                  // combine M 0..7 with 8..15
        if (lane < 16)
            Ssum[(long)blk * 256 + nbase + j * 16 + lane] = s;
    }
}

// ---------------------------------------------------------------------------
// out0[b,h,w,c] = S0[b,h/4,w/4,c] + S1[b,h,w,c]        (h,w in 32x32)
// ---------------------------------------------------------------------------
__global__ __launch_bounds__(256)
void out0_kernel(const float* __restrict__ S0, const float* __restrict__ S1,
                 float* __restrict__ out0) {
    const long i = (long)blockIdx.x * blockDim.x + threadIdx.x;  // float4 index
    if (i >= 8L * 32 * 32 * 64) return;
    const long c4 = i & 63;
    const long w  = (i >> 6) & 31;
    const long h  = (i >> 11) & 31;
    const long b  = i >> 16;
    const long p  = ((b * 8 + (h >> 2)) * 8 + (w >> 2)) * 64 + c4;
    const float4 a = ((const float4*)S0)[p];
    const float4 s = ((const float4*)S1)[i];
    float4 o; o.x = a.x + s.x; o.y = a.y + s.y; o.z = a.z + s.z; o.w = a.w + s.w;
    ((float4*)out0)[i] = o;
}

// ---------------------------------------------------------------------------
// out1[b,h,w,c] = S1[b,h/4,w/4,c] + v0[b,h/4,w/4,c]    (h,w in 128x128)
// ---------------------------------------------------------------------------
__global__ __launch_bounds__(256)
void out1_kernel(const float* __restrict__ S1, const float* __restrict__ V0,
                 float* __restrict__ out1) {
    const long i = (long)blockIdx.x * blockDim.x + threadIdx.x;  // float4 index
    if (i >= 8L * 128 * 128 * 64) return;
    const long c4 = i & 63;
    const long w  = (i >> 6) & 127;
    const long h  = (i >> 13) & 127;
    const long b  = i >> 20;
    const long p  = ((b * 32 + (h >> 2)) * 32 + (w >> 2)) * 64 + c4;
    const float4 s = ((const float4*)S1)[p];
    const float4 v = ((const float4*)V0)[p];
    float4 o; o.x = s.x + v.x; o.y = s.y + v.y; o.z = s.z + v.z; o.w = s.w + v.w;
    ((float4*)out1)[i] = o;
}

// ---------------------------------------------------------------------------
extern "C" void kernel_launch(void* const* d_in, const int* in_sizes, int n_in,
                              void* d_out, int out_size, void* d_ws, size_t ws_size,
                              hipStream_t stream) {
    (void)in_sizes; (void)n_in; (void)out_size; (void)ws_size;

    const float* x0 = (const float*)d_in[0];   // (8,32,32,256)
    const float* x1 = (const float*)d_in[1];   // (8,128,128,256)
    const float* W0 = (const float*)d_in[2];   // (256,768)
    const float* W1 = (const float*)d_in[3];   // (256,768)

    char* ws = (char*)d_ws;
    bf16_t* Wb0 = (bf16_t*)(ws);                                   // 131072 B
    bf16_t* Wb1 = (bf16_t*)(ws + 131072);                          // 131072 B
    float*  V0  = (float*)(ws + 262144);                           // 8 MiB
    float*  S0  = (float*)(ws + 262144 + 8388608);                 // 512 KiB
    float*  S1  = (float*)(ws + 262144 + 8388608 + 524288);        // 8 MiB

    float* out0 = (float*)d_out;
    float* out1 = out0 + 8L * 32 * 32 * 256;

    convert_w_kernel<<<256, 32, 0, stream>>>(W0, W1, Wb0, Wb1);
    gemm_v_block_kernel<<<512, 128, 0, stream>>>(x0, Wb0, V0, S0, 32);
    gemm_v_block_kernel<<<8192, 128, 0, stream>>>(x1, Wb1, nullptr, S1, 128);
    out0_kernel<<<2048, 256, 0, stream>>>(S0, S1, out0);
    out1_kernel<<<32768, 256, 0, stream>>>(S1, V0, out1);
}
